// GATTemporalModel_7928509628483
// MI455X (gfx1250) — matrix operand order repaired and checked by reading, not compiled, not run on hardware
//
#include <hip/hip_runtime.h>

#define N_NODES   25600
#define E_EDGES   204800
#define ETOT      (E_EDGES + N_NODES)   // with self loops
#define IN_CH     6
#define HID       64
#define HEADS     4
#define F1        256                   // HEADS*HID
#define GBATCH    512
#define TSEQ      50
#define OUTC      30
#define NEG_SLOPE 0.2f
#define BN_INVS   0.9999950000374997f   // 1/sqrt(1+1e-5)

typedef __attribute__((ext_vector_type(16))) _Float16 v16h;
typedef __attribute__((ext_vector_type(8)))  float    v8f;

// ---------- helpers ----------
__device__ __forceinline__ float lrelu(float v) { return v > 0.f ? v : NEG_SLOPE * v; }
__device__ __forceinline__ float eluf (float v) { return v > 0.f ? v : (expf(v) - 1.f); }
__device__ __forceinline__ float sigf (float v) { return 1.f / (1.f + expf(-v)); }

__device__ __forceinline__ void atomicMaxF(float* addr, float v) {
    // valid when addr initialized to -inf: ordered max via sign-split integer atomics
    if (v >= 0.f) atomicMax((int*)addr, __float_as_int(v));
    else          atomicMin((unsigned int*)addr, (unsigned int)__float_as_int(v));
}

// ---------- generic fill ----------
__global__ void k_fill(float* p, long n, float v) {
    long i = (long)blockIdx.x * blockDim.x + threadIdx.x;
    if (i < n) p[i] = v;
}

// ---------- format fp32 weights directly into WMMA B-fragment layout ----------
// Logical B is (K x Nc). Fragment layout: for k-step ks (32 rows) and n-tile nt,
// lane L holds halves j=0..15 with  col = nt*16 + (L&15),  k = ks*32 + (L<16?0:16) + j,
// stored contiguously: Bf[(((ks*NT)+nt)*32 + L)*16 + j]  -> per-lane v16h load.
// trans==0: B[k][col] = W[k*Nc+col]   (W stored K x Nc)
// trans==1: B[k][col] = W[col*K+k]    (W stored Nc x K, e.g. Wih/Whh)
__global__ void k_fmtB(const float* __restrict__ W, _Float16* __restrict__ Bf,
                       int K, int Nc, int trans) {
    int id = blockIdx.x * blockDim.x + threadIdx.x;
    if (id >= K * Nc) return;
    int j    = id & 15;
    int lane = (id >> 4) & 31;
    int rest = id >> 9;
    int NT   = Nc >> 4;
    int nt   = rest % NT;
    int ks   = rest / NT;
    int col  = nt * 16 + (lane & 15);
    int k    = ks * 32 + ((lane < 16) ? 0 : 16) + j;
    float v  = trans ? W[col * K + k] : W[k * Nc + col];
    Bf[id] = (_Float16)v;
}

// ---------- WMMA f16->f32 matmul: D(MxNc) = A(MxK) * Bf [+ Cinit] ----------
// K, Nc compile-time so the k-loop fully unrolls; each wave owns NTW n-tiles
// sharing one A fragment (NTW independent back-to-back v_wmma per k-step).
template<int K, int Nc, int NTW>
__global__ void k_wmma_mm(const _Float16* __restrict__ A, const _Float16* __restrict__ Bf,
                          const float* __restrict__ Cinit, float* __restrict__ D, int M) {
    constexpr int NT = Nc / 16;       // n-tiles total
    constexpr int NG = NT / NTW;      // n-tile groups per row-tile
    int wave = (blockIdx.x * blockDim.x + threadIdx.x) >> 5;
    int lane = threadIdx.x & 31;
    int mt = wave / NG;
    int ng = wave - mt * NG;
    if (mt * 16 >= M) return;         // whole-wave uniform exit (EXEC all-1 at wmma)

    int r     = lane & 15;
    int rowA  = mt * 16 + r;                      // A: lane holds row (lane&15)
    int kbA   = (lane < 16) ? 0 : 16;             // contiguous k: hi lanes hold k 16..31 (B same)
    int rowD0 = mt * 16 + ((lane < 16) ? 0 : 8);  // C/D: VGPR i -> row i (+8 hi lanes)

    v8f acc[NTW];
#pragma unroll
    for (int w = 0; w < NTW; ++w) {
        int colB = (ng * NTW + w) * 16 + r;
#pragma unroll
        for (int i = 0; i < 8; ++i)
            acc[w][i] = Cinit ? Cinit[(size_t)(rowD0 + i) * Nc + colB] : 0.f;
    }

#pragma unroll
    for (int ks = 0; ks < K / 32; ++ks) {
        const _Float16* Ar = A + (size_t)rowA * K + ks * 32;
        __builtin_prefetch(Ar + 64, 0, 1);        // global_prefetch_b8
        v16h a;
#pragma unroll
        for (int j = 0; j < 8; ++j) {             // two contiguous 16B chunks -> b128 x2
            a[j]     = Ar[kbA + j];
            a[8 + j] = Ar[kbA + 8 + j];
        }
#pragma unroll
        for (int w = 0; w < NTW; ++w) {
            const v16h* bp = (const v16h*)(Bf +
                (((size_t)ks * NT + ng * NTW + w) * 32 + lane) * 16);
            acc[w] = __builtin_amdgcn_wmma_f32_16x16x32_f16(false, a, false, *bp,
                                                            (short)0, acc[w], false, false);
        }
    }

#pragma unroll
    for (int w = 0; w < NTW; ++w) {
        int colB = (ng * NTW + w) * 16 + r;
#pragma unroll
        for (int i = 0; i < 8; ++i)
            D[(size_t)(rowD0 + i) * Nc + colB] = acc[w][i];
    }
}

// ---------- GAT layer 1: h1raw = x @ W1 (N x 256) ----------
__global__ void k_gat1(const float* __restrict__ x, const float* __restrict__ W1,
                       float* __restrict__ h1raw) {
    int n = blockIdx.x, c = threadIdx.x;
    float acc = 0.f;
#pragma unroll
    for (int k = 0; k < IN_CH; ++k)
        acc += x[n * IN_CH + k] * W1[k * F1 + c];
    h1raw[(size_t)n * F1 + c] = acc;
}

// per-node attention scores: s[n,h] = sum_c h[n,h,c]*a[h,c]
__global__ void k_scores1(const float* __restrict__ h, const float* __restrict__ asrc,
                          const float* __restrict__ adst, float* ssrc, float* sdst) {
    int id = blockIdx.x * blockDim.x + threadIdx.x;
    if (id >= N_NODES * HEADS) return;
    int n = id / HEADS, hh = id % HEADS;
    float s0 = 0.f, s1 = 0.f;
    const float* hp = h + (size_t)n * F1 + hh * HID;
    for (int c = 0; c < HID; ++c) {
        s0 += hp[c] * asrc[hh * HID + c];
        s1 += hp[c] * adst[hh * HID + c];
    }
    ssrc[id] = s0; sdst[id] = s1;
}

__global__ void k_edge_max1(const int* __restrict__ EI, const float* ssrc,
                            const float* sdst, float* m, float* esc) {
    int id = blockIdx.x * blockDim.x + threadIdx.x;
    if (id >= ETOT * HEADS) return;
    int e = id / HEADS, hh = id % HEADS;
    int src = (e < E_EDGES) ? EI[e] : (e - E_EDGES);
    int dst = (e < E_EDGES) ? EI[E_EDGES + e] : (e - E_EDGES);
    float sc = lrelu(ssrc[src * HEADS + hh] + sdst[dst * HEADS + hh]);
    esc[id] = sc;
    atomicMaxF(&m[dst * HEADS + hh], sc);
}

__global__ void k_edge_exp1(const int* __restrict__ EI, const float* m,
                            float* esc, float* denom) {
    int id = blockIdx.x * blockDim.x + threadIdx.x;
    if (id >= ETOT * HEADS) return;
    int e = id / HEADS, hh = id % HEADS;
    int dst = (e < E_EDGES) ? EI[E_EDGES + e] : (e - E_EDGES);
    float ex = expf(esc[id] - m[dst * HEADS + hh]);
    esc[id] = ex;
    atomicAdd(&denom[dst * HEADS + hh], ex);
}

__global__ void k_agg1(const int* __restrict__ EI, const float* __restrict__ esc,
                       const float* __restrict__ h, float* __restrict__ out) {
    int e = blockIdx.x, c = threadIdx.x;          // 256 threads
    int src = (e < E_EDGES) ? EI[e] : (e - E_EDGES);
    int dst = (e < E_EDGES) ? EI[E_EDGES + e] : (e - E_EDGES);
    float coef = esc[e * HEADS + (c >> 6)];
    atomicAdd(&out[(size_t)dst * F1 + c], coef * h[(size_t)src * F1 + c]);
}

__global__ void k_fin1(const float* __restrict__ out, const float* denom,
                       const float* b1, const float* g1, const float* be1,
                       _Float16* __restrict__ h1h) {
    int n = blockIdx.x, c = threadIdx.x;
    size_t id = (size_t)n * F1 + c;
    float v = out[id] / denom[n * HEADS + (c >> 6)] + b1[c];
    v = g1[c] * (v * BN_INVS) + be1[c];
    h1h[id] = (_Float16)eluf(v);
}

// ---------- GAT layer 2 helpers (1 head, 64 ch) ----------
__global__ void k_scores2(const float* __restrict__ h, const float* asrc,
                          const float* adst, float* ssrc, float* sdst) {
    int n = blockIdx.x * blockDim.x + threadIdx.x;
    if (n >= N_NODES) return;
    float s0 = 0.f, s1 = 0.f;
    const float* hp = h + (size_t)n * HID;
    for (int c = 0; c < HID; ++c) { s0 += hp[c] * asrc[c]; s1 += hp[c] * adst[c]; }
    ssrc[n] = s0; sdst[n] = s1;
}

__global__ void k_edge_max2(const int* EI, const float* ssrc, const float* sdst,
                            float* m, float* esc) {
    int e = blockIdx.x * blockDim.x + threadIdx.x;
    if (e >= ETOT) return;
    int src = (e < E_EDGES) ? EI[e] : (e - E_EDGES);
    int dst = (e < E_EDGES) ? EI[E_EDGES + e] : (e - E_EDGES);
    float sc = lrelu(ssrc[src] + sdst[dst]);
    esc[e] = sc;
    atomicMaxF(&m[dst], sc);
}

__global__ void k_edge_exp2(const int* EI, const float* m, float* esc, float* denom) {
    int e = blockIdx.x * blockDim.x + threadIdx.x;
    if (e >= ETOT) return;
    int dst = (e < E_EDGES) ? EI[E_EDGES + e] : (e - E_EDGES);
    float ex = expf(esc[e] - m[dst]);
    esc[e] = ex;
    atomicAdd(&denom[dst], ex);
}

__global__ void k_agg2(const int* __restrict__ EI, const float* __restrict__ esc,
                       const float* __restrict__ h, float* __restrict__ out) {
    int e = blockIdx.x, c = threadIdx.x;          // 64 threads
    int src = (e < E_EDGES) ? EI[e] : (e - E_EDGES);
    int dst = (e < E_EDGES) ? EI[E_EDGES + e] : (e - E_EDGES);
    atomicAdd(&out[(size_t)dst * HID + c], esc[e] * h[(size_t)src * HID + c]);
}

// finalize layer 2 + reshape node n=(g*T+t) into LSTM A-matrix row (t*G+g), fp16
__global__ void k_fin2(const float* __restrict__ out, const float* denom,
                       const float* b2, const float* g2, const float* be2,
                       _Float16* __restrict__ xh) {
    int id = blockIdx.x * blockDim.x + threadIdx.x;
    if (id >= N_NODES * HID) return;
    int n = id >> 6, c = id & 63;
    float v = out[id] / denom[n] + b2[c];
    v = g2[c] * (v * BN_INVS) + be2[c];
    v = eluf(v);
    int g = n / TSEQ, t = n % TSEQ;
    xh[((size_t)t * GBATCH + g) * HID + c] = (_Float16)v;
}

// ---------- LSTM pointwise: gates(512x256) -> h,c update ----------
__global__ void k_lstm_pw(const float* __restrict__ gates, const float* bih,
                          const float* bhh, float* __restrict__ hst,
                          float* __restrict__ cst, _Float16* __restrict__ hh16) {
    int id = blockIdx.x * blockDim.x + threadIdx.x;
    if (id >= GBATCH * HID) return;
    int g = id >> 6, c = id & 63;
    const float* gp = gates + (size_t)g * (4 * HID);
    float i  = sigf (gp[c]             + bih[c]             + bhh[c]);
    float f  = sigf (gp[HID + c]       + bih[HID + c]       + bhh[HID + c]);
    float gg = tanhf(gp[2 * HID + c]   + bih[2 * HID + c]   + bhh[2 * HID + c]);
    float o  = sigf (gp[3 * HID + c]   + bih[3 * HID + c]   + bhh[3 * HID + c]);
    float cn = f * cst[id] + i * gg;
    float hn = o * tanhf(cn);
    cst[id] = cn; hst[id] = hn; hh16[id] = (_Float16)hn;
}

// ---------- FC head ----------
__global__ void k_fc(const float* __restrict__ h, const float* __restrict__ W,
                     const float* b, float* __restrict__ out) {
    int id = blockIdx.x * blockDim.x + threadIdx.x;
    if (id >= GBATCH * OUTC) return;
    int g = id / OUTC, o = id % OUTC;
    float acc = b[o];
    for (int k = 0; k < HID; ++k) acc += h[g * HID + k] * W[k * OUTC + o];
    out[id] = acc;
}

extern "C" void kernel_launch(void* const* d_in, const int* in_sizes, int n_in,
                              void* d_out, int out_size, void* d_ws, size_t ws_size,
                              hipStream_t stream) {
    (void)in_sizes; (void)n_in; (void)out_size; (void)ws_size;
    const float* x    = (const float*)d_in[0];
    const int*   EI   = (const int*)  d_in[1];
    const float* W1   = (const float*)d_in[4];
    const float* as1  = (const float*)d_in[5];
    const float* ad1  = (const float*)d_in[6];
    const float* b1   = (const float*)d_in[7];
    const float* g1   = (const float*)d_in[8];
    const float* be1  = (const float*)d_in[9];
    const float* W2   = (const float*)d_in[10];
    const float* as2  = (const float*)d_in[11];
    const float* ad2  = (const float*)d_in[12];
    const float* b2   = (const float*)d_in[13];
    const float* g2   = (const float*)d_in[14];
    const float* be2  = (const float*)d_in[15];
    const float* Wih  = (const float*)d_in[16];
    const float* Whh  = (const float*)d_in[17];
    const float* bih  = (const float*)d_in[18];
    const float* bhh  = (const float*)d_in[19];
    const float* Wfc  = (const float*)d_in[20];
    const float* bfc  = (const float*)d_in[21];
    float* out = (float*)d_out;

    // ---- carve workspace ----
    size_t off = 0;
    auto carve = [&](size_t bytes) -> void* {
        void* p = (char*)d_ws + off;
        off += (bytes + 255) & ~(size_t)255;
        return p;
    };
    float*    h1raw  = (float*)   carve((size_t)N_NODES * F1 * 4);
    _Float16* h1h    = (_Float16*)carve((size_t)N_NODES * F1 * 2);
    float*    ssrc1  = (float*)   carve((size_t)N_NODES * HEADS * 4);
    float*    sdst1  = (float*)   carve((size_t)N_NODES * HEADS * 4);
    float*    m1     = (float*)   carve((size_t)N_NODES * HEADS * 4);
    float*    den1   = (float*)   carve((size_t)N_NODES * HEADS * 4);
    float*    esc1   = (float*)   carve((size_t)ETOT * HEADS * 4);
    float*    out1   = (float*)   carve((size_t)N_NODES * F1 * 4);
    _Float16* W2f    = (_Float16*)carve((size_t)F1 * HID * 2);       // B fragments 256x64
    float*    h2raw  = (float*)   carve((size_t)N_NODES * HID * 4);
    float*    ssrc2  = (float*)   carve((size_t)N_NODES * 4);
    float*    sdst2  = (float*)   carve((size_t)N_NODES * 4);
    float*    m2     = (float*)   carve((size_t)N_NODES * 4);
    float*    den2   = (float*)   carve((size_t)N_NODES * 4);
    float*    esc2   = (float*)   carve((size_t)ETOT * 4);
    float*    out2   = (float*)   carve((size_t)N_NODES * HID * 4);
    _Float16* xh     = (_Float16*)carve((size_t)N_NODES * HID * 2);
    _Float16* Wihf   = (_Float16*)carve((size_t)HID * 4 * HID * 2);  // B fragments 64x256
    _Float16* Whhf   = (_Float16*)carve((size_t)HID * 4 * HID * 2);  // B fragments 64x256
    float*    XG     = (float*)   carve((size_t)N_NODES * (4 * HID) * 4);
    float*    gates  = (float*)   carve((size_t)GBATCH * (4 * HID) * 4);
    float*    hst    = (float*)   carve((size_t)GBATCH * HID * 4);
    float*    cst    = (float*)   carve((size_t)GBATCH * HID * 4);
    _Float16* hh16   = (_Float16*)carve((size_t)GBATCH * HID * 2);

    const float NEG_INF = -__builtin_huge_valf();

    // ---- init accumulators / state (every call: ws is not re-zeroed by harness) ----
    k_fill<<<(N_NODES * HEADS + 255) / 256, 256, 0, stream>>>(m1,  N_NODES * HEADS, NEG_INF);
    k_fill<<<(N_NODES * HEADS + 255) / 256, 256, 0, stream>>>(den1, N_NODES * HEADS, 0.f);
    k_fill<<<((long)N_NODES * F1 + 255) / 256, 256, 0, stream>>>(out1, (long)N_NODES * F1, 0.f);
    k_fill<<<(N_NODES + 255) / 256, 256, 0, stream>>>(m2,  N_NODES, NEG_INF);
    k_fill<<<(N_NODES + 255) / 256, 256, 0, stream>>>(den2, N_NODES, 0.f);
    k_fill<<<((long)N_NODES * HID + 255) / 256, 256, 0, stream>>>(out2, (long)N_NODES * HID, 0.f);
    k_fill<<<(GBATCH * HID + 255) / 256, 256, 0, stream>>>(hst, GBATCH * HID, 0.f);
    k_fill<<<(GBATCH * HID + 255) / 256, 256, 0, stream>>>(cst, GBATCH * HID, 0.f);
    k_fill<<<(GBATCH * HID / 2 + 255) / 256, 256, 0, stream>>>((float*)hh16, GBATCH * HID / 2, 0.f);

    // ---- weight -> B-fragment conversions ----
    k_fmtB<<<(F1 * HID + 255) / 256, 256, 0, stream>>>(W2,  W2f,  F1,  HID,     0); // 256x64
    k_fmtB<<<(HID * 4 * HID + 255) / 256, 256, 0, stream>>>(Wih, Wihf, HID, 4 * HID, 1); // 64x256
    k_fmtB<<<(HID * 4 * HID + 255) / 256, 256, 0, stream>>>(Whh, Whhf, HID, 4 * HID, 1); // 64x256

    // ---- GAT layer 1 ----
    k_gat1<<<N_NODES, F1, 0, stream>>>(x, W1, h1raw);
    k_scores1<<<(N_NODES * HEADS + 255) / 256, 256, 0, stream>>>(h1raw, as1, ad1, ssrc1, sdst1);
    k_edge_max1<<<(ETOT * HEADS + 255) / 256, 256, 0, stream>>>(EI, ssrc1, sdst1, m1, esc1);
    k_edge_exp1<<<(ETOT * HEADS + 255) / 256, 256, 0, stream>>>(EI, m1, esc1, den1);
    k_agg1<<<ETOT, F1, 0, stream>>>(EI, esc1, h1raw, out1);
    k_fin1<<<N_NODES, F1, 0, stream>>>(out1, den1, b1, g1, be1, h1h);

    // ---- GAT layer 2 transform: h2raw = h1 @ W2  (WMMA: M=25600 Nc=64 K=256) ----
    k_wmma_mm<F1, HID, 4><<<(N_NODES / 16) * (HID / 64) / 8, 256, 0, stream>>>(
        h1h, W2f, nullptr, h2raw, N_NODES);
    k_scores2<<<(N_NODES + 255) / 256, 256, 0, stream>>>(h2raw, as2, ad2, ssrc2, sdst2);
    k_edge_max2<<<(ETOT + 255) / 256, 256, 0, stream>>>(EI, ssrc2, sdst2, m2, esc2);
    k_edge_exp2<<<(ETOT + 255) / 256, 256, 0, stream>>>(EI, m2, esc2, den2);
    k_agg2<<<ETOT, HID, 0, stream>>>(EI, esc2, h2raw, out2);
    k_fin2<<<((long)N_NODES * HID + 255) / 256, 256, 0, stream>>>(out2, den2, b2, g2, be2, xh);

    // ---- LSTM input gates all timesteps: XG = xh @ Wih^T (WMMA: M=25600 Nc=256 K=64) ----
    k_wmma_mm<HID, 4 * HID, 4><<<(N_NODES / 16) * (4 * HID / 64) / 8, 256, 0, stream>>>(
        xh, Wihf, nullptr, XG, N_NODES);

    // ---- LSTM recurrence (WMMA per step, C-init = XG slice) ----
    for (int t = 0; t < TSEQ; ++t) {
        k_wmma_mm<HID, 4 * HID, 4><<<(GBATCH / 16) * (4 * HID / 64) / 8, 256, 0, stream>>>(
            hh16, Whhf, XG + (size_t)t * GBATCH * 4 * HID, gates, GBATCH);
        k_lstm_pw<<<(GBATCH * HID + 255) / 256, 256, 0, stream>>>(
            gates, bih, bhh, hst, cst, hh16);
    }

    // ---- FC head ----
    k_fc<<<(GBATCH * OUTC + 255) / 256, 256, 0, stream>>>(hst, Wfc, bfc, out);
}
